// GNNEncoder_16595753632014
// MI455X (gfx1250) — matrix operand order repaired and checked by reading, not compiled, not run on hardware
//
#include <hip/hip_runtime.h>
#include <hip/hip_bf16.h>

#define N_NODES   100000
#define N_EDGES   600000
#define NODE_DIM  64
#define EDGE_DIM  32
#define HID       128
#define N_GRAPHS  256
#define LN_EPS    1e-5f

typedef __attribute__((ext_vector_type(8)))  float  v8f;
typedef __bf16 bf16_t;
typedef __attribute__((ext_vector_type(16))) __bf16 v16bf;

union AFrag { v16bf v; uint4 u[2]; bf16_t e[16]; };

// ---- WMMA fragment loaders -------------------------------------------------
// A-matrix 16x32 bf16 layout (ISA 7.12.2): lane group 0 (lanes 0-15) holds
// K = kb+0..7 (VGPR0-3) and kb+16..23 (VGPR4-7); group 1 holds kb+8..15 and
// kb+24..31. Each half is 8 contiguous bf16 = one 16-byte load.
__device__ inline v16bf load_frag_bf(const bf16_t* base, int kb, int grp) {
  AFrag f;
  f.u[0] = *(const uint4*)(base + kb + grp * 8);
  f.u[1] = *(const uint4*)(base + kb + 16 + grp * 8);
  return f.v;
}
__device__ inline v16bf load_frag_f32(const float* base, int kb, int grp) {
  AFrag f;
  const float* p0 = base + kb + grp * 8;
  const float* p1 = base + kb + 16 + grp * 8;
#pragma unroll
  for (int i = 0; i < 8; i++) { f.e[i] = (bf16_t)p0[i]; f.e[8 + i] = (bf16_t)p1[i]; }
  return f.v;
}
// B-matrix 32x16 bf16: same K split as A (group g: kb+8g..+7, kb+16+8g..+7),
// N = lane&15. Weights stored transposed Wt[n][k].
__device__ inline v16bf load_b_frag(const bf16_t* Wt, int K, int n, int kb, int grp) {
  AFrag f;
  const bf16_t* p = Wt + (size_t)n * K + kb + grp * 8;
  f.u[0] = *(const uint4*)(p);
  f.u[1] = *(const uint4*)(p + 16);
  return f.v;
}
__device__ inline v8f wmma_bf16(v16bf a, v16bf b, v8f c) {
  return __builtin_amdgcn_wmma_f32_16x16x32_bf16(false, a, false, b, (short)0, c,
                                                 false, false);
}
__device__ inline float gelu_exact(float x) {
  return 0.5f * x * (1.0f + erff(x * 0.70710678118654752f));
}
__device__ inline void atomic_add_f32(float* p, float v) {
  __hip_atomic_fetch_add(p, v, __ATOMIC_RELAXED, __HIP_MEMORY_SCOPE_AGENT);
}
#define WAVE_FENCE() asm volatile("" ::: "memory")

// ---- weight transpose + fp32->bf16 ----------------------------------------
__global__ void k_wt(const float* __restrict__ W, bf16_t* __restrict__ Wt,
                     int fin, int fout) {
  int i = blockIdx.x * blockDim.x + threadIdx.x;
  if (i < fin * fout) {
    int o = i / fin, k = i % fin;
    Wt[i] = (bf16_t)W[(size_t)k * fout + o];
  }
}

// ---- fused  y = gelu(LN(x @ W + b))  projection (node/edge encoder) --------
template <int FIN>
__global__ __launch_bounds__(256) void k_proj(
    const float* __restrict__ feats, int nrows, const bf16_t* __restrict__ Wt,
    const float* __restrict__ bias, const float* __restrict__ g,
    const float* __restrict__ be, float* __restrict__ out_f32,
    bf16_t* __restrict__ out_bf) {
  constexpr int KS = FIN / 32;
  __shared__ float smem[8 * 16 * HID];
  const int wid = threadIdx.x >> 5, lane = threadIdx.x & 31;
  const int grp = lane >> 4, l15 = lane & 15;
  const int tile = blockIdx.x * 8 + wid;
  if (tile * 16 >= nrows) return;
  float* tb = smem + wid * (16 * HID);

  const float* arow = feats + (size_t)(tile * 16 + l15) * FIN;
  v16bf af[KS];
#pragma unroll
  for (int ks = 0; ks < KS; ks++) af[ks] = load_frag_f32(arow, ks * 32, grp);

#pragma unroll
  for (int nt = 0; nt < 8; nt++) {
    v8f acc = {0.f, 0.f, 0.f, 0.f, 0.f, 0.f, 0.f, 0.f};
#pragma unroll
    for (int ks = 0; ks < KS; ks++) {
      v16bf bf = load_b_frag(Wt, FIN, nt * 16 + l15, ks * 32, grp);
      acc = wmma_bf16(af[ks], bf, acc);
    }
    float bc = bias[nt * 16 + l15];
#pragma unroll
    for (int j = 0; j < 8; j++)
      tb[(j + grp * 8) * HID + nt * 16 + l15] = acc[j] + bc;  // C/D layout
  }
  WAVE_FENCE();
  // LayerNorm + GELU: lane owns half of row l15 (cols grp*64..+63)
  float v[64], s = 0.f, ss = 0.f;
#pragma unroll
  for (int c = 0; c < 64; c++) {
    float t = tb[l15 * HID + grp * 64 + c];
    v[c] = t; s += t; ss += t * t;
  }
  s += __shfl_xor(s, 16); ss += __shfl_xor(ss, 16);
  float mu = s * (1.f / HID);
  float rstd = rsqrtf(ss * (1.f / HID) - mu * mu + LN_EPS);
  const size_t orow = (size_t)(tile * 16 + l15) * HID;
#pragma unroll
  for (int c = 0; c < 64; c++) {
    int col = grp * 64 + c;
    float y = gelu_exact((v[c] - mu) * rstd * g[col] + be[col]);
    if (out_f32) out_f32[orow + col] = y;
    out_bf[orow + col] = (bf16_t)y;
  }
}

// ---- fused message layer: gather -> GEMM1 -> LN/GELU -> GEMM2 -> LN/GELU ->
// ---- scatter-add (fp32 atomics into L2-resident accumulator) ---------------
__global__ __launch_bounds__(256) void k_msg(
    const bf16_t* __restrict__ x_bf, const bf16_t* __restrict__ e_bf,
    const int* __restrict__ row_idx, const int* __restrict__ col_idx,
    const bf16_t* __restrict__ W1t, const float* __restrict__ b1,
    const float* __restrict__ g1, const float* __restrict__ be1,
    const bf16_t* __restrict__ W2t, const float* __restrict__ b2,
    const float* __restrict__ g2, const float* __restrict__ be2,
    float* __restrict__ msum) {
  __shared__ float smem[8 * 16 * HID];
  const int wid = threadIdx.x >> 5, lane = threadIdx.x & 31;
  const int grp = lane >> 4, l15 = lane & 15;
  const int tile = blockIdx.x * 8 + wid;
  if (tile * 16 >= N_EDGES) return;
  float* tb = smem + wid * (16 * HID);
  bf16_t* tbh = (bf16_t*)tb;

  const int e = tile * 16 + l15;
  const int r = row_idx[e];
  const bf16_t* xr = x_bf + (size_t)r * HID;   // gathered source-node row
  const bf16_t* er = e_bf + (size_t)e * HID;   // edge feature row
  v16bf a1[8];
#pragma unroll
  for (int ks = 0; ks < 4; ks++) a1[ks] = load_frag_bf(xr, ks * 32, grp);
#pragma unroll
  for (int ks = 0; ks < 4; ks++) a1[4 + ks] = load_frag_bf(er, ks * 32, grp);

  // GEMM1: [16,256] x [256,128]
#pragma unroll
  for (int nt = 0; nt < 8; nt++) {
    v8f acc = {0.f, 0.f, 0.f, 0.f, 0.f, 0.f, 0.f, 0.f};
#pragma unroll
    for (int ks = 0; ks < 8; ks++) {
      v16bf bfrag = load_b_frag(W1t, 2 * HID, nt * 16 + l15, ks * 32, grp);
      acc = wmma_bf16(a1[ks], bfrag, acc);
    }
    float bc = b1[nt * 16 + l15];
#pragma unroll
    for (int j = 0; j < 8; j++)
      tb[(j + grp * 8) * HID + nt * 16 + l15] = acc[j] + bc;
  }
  WAVE_FENCE();
  // LN1 + GELU, rewrite tile as bf16 (all per-wave LDS reads precede writes)
  {
    float v[64], s = 0.f, ss = 0.f;
#pragma unroll
    for (int c = 0; c < 64; c++) {
      float t = tb[l15 * HID + grp * 64 + c];
      v[c] = t; s += t; ss += t * t;
    }
    s += __shfl_xor(s, 16); ss += __shfl_xor(ss, 16);
    float mu = s * (1.f / HID);
    float rstd = rsqrtf(ss * (1.f / HID) - mu * mu + LN_EPS);
    WAVE_FENCE();
#pragma unroll
    for (int c = 0; c < 64; c++) {
      int col = grp * 64 + c;
      tbh[l15 * HID + col] =
          (bf16_t)gelu_exact((v[c] - mu) * rstd * g1[col] + be1[col]);
    }
  }
  WAVE_FENCE();
  // GEMM2: [16,128] x [128,128]; preload A from LDS before overwriting tile
  v16bf a2[4];
#pragma unroll
  for (int ks = 0; ks < 4; ks++)
    a2[ks] = load_frag_bf(tbh + l15 * HID, ks * 32, grp);
#pragma unroll
  for (int nt = 0; nt < 8; nt++) {
    v8f acc = {0.f, 0.f, 0.f, 0.f, 0.f, 0.f, 0.f, 0.f};
#pragma unroll
    for (int ks = 0; ks < 4; ks++) {
      v16bf bfrag = load_b_frag(W2t, HID, nt * 16 + l15, ks * 32, grp);
      acc = wmma_bf16(a2[ks], bfrag, acc);
    }
    float bc = b2[nt * 16 + l15];
#pragma unroll
    for (int j = 0; j < 8; j++)
      tb[(j + grp * 8) * HID + nt * 16 + l15] = acc[j] + bc;
  }
  WAVE_FENCE();
  // LN2 + GELU + scatter-mean numerator (atomic adds; denominator precomputed)
  {
    float v[64], s = 0.f, ss = 0.f;
#pragma unroll
    for (int c = 0; c < 64; c++) {
      float t = tb[l15 * HID + grp * 64 + c];
      v[c] = t; s += t; ss += t * t;
    }
    s += __shfl_xor(s, 16); ss += __shfl_xor(ss, 16);
    float mu = s * (1.f / HID);
    float rstd = rsqrtf(ss * (1.f / HID) - mu * mu + LN_EPS);
    const int dst = col_idx[e];
    float* mrow = msum + (size_t)dst * HID;
#pragma unroll
    for (int c = 0; c < 64; c++) {
      int col = grp * 64 + c;
      float y = gelu_exact((v[c] - mu) * rstd * g2[col] + be2[col]);
      atomic_add_f32(mrow + col, y);
    }
  }
}

// ---- helpers ---------------------------------------------------------------
__global__ void k_degree(const int* __restrict__ col_idx, float* __restrict__ cnt) {
  int i = blockIdx.x * blockDim.x + threadIdx.x;
  if (i < N_EDGES) atomic_add_f32(&cnt[col_idx[i]], 1.0f);
}

__global__ void k_update(float* __restrict__ x, bf16_t* __restrict__ x_bf,
                         const float* __restrict__ msum,
                         const float* __restrict__ cnt) {
  int i = blockIdx.x * blockDim.x + threadIdx.x;
  if (i < N_NODES * HID) {
    float nx = x[i] + msum[i] / fmaxf(cnt[i >> 7], 1.0f);
    x[i] = nx;
    x_bf[i] = (bf16_t)nx;
  }
}

__global__ __launch_bounds__(256) void k_final(
    const float* __restrict__ x, const int* __restrict__ batch,
    const float* __restrict__ g, const float* __restrict__ be,
    float* __restrict__ gsum, float* __restrict__ gcnt) {
  const int wid = threadIdx.x >> 5, lane = threadIdx.x & 31;
  const int r = blockIdx.x * 8 + wid;  // one row per wave
  if (r >= N_NODES) return;
  const float* xr = x + (size_t)r * HID;
  float v[4], s = 0.f, ss = 0.f;
#pragma unroll
  for (int k = 0; k < 4; k++) {
    float t = xr[lane + 32 * k];
    v[k] = t; s += t; ss += t * t;
  }
#pragma unroll
  for (int o = 16; o > 0; o >>= 1) { s += __shfl_xor(s, o); ss += __shfl_xor(ss, o); }
  float mu = s * (1.f / HID);
  float rstd = rsqrtf(ss * (1.f / HID) - mu * mu + LN_EPS);
  const int gi = batch[r];
  float* gr = gsum + (size_t)gi * HID;
#pragma unroll
  for (int k = 0; k < 4; k++) {
    int col = lane + 32 * k;
    atomic_add_f32(gr + col, (v[k] - mu) * rstd * g[col] + be[col]);
  }
  if (lane == 0) atomic_add_f32(&gcnt[gi], 1.0f);
}

__global__ void k_out(const float* __restrict__ gsum,
                      const float* __restrict__ gcnt, float* __restrict__ out) {
  int i = blockIdx.x * blockDim.x + threadIdx.x;
  if (i < N_GRAPHS * HID) out[i] = gsum[i] / fmaxf(gcnt[i >> 7], 1.0f);
}

// ---- host side -------------------------------------------------------------
extern "C" void kernel_launch(void* const* d_in, const int* in_sizes, int n_in,
                              void* d_out, int out_size, void* d_ws,
                              size_t ws_size, hipStream_t stream) {
  (void)in_sizes; (void)n_in; (void)out_size; (void)ws_size;
  // inputs
  const float* node_feats = (const float*)d_in[0];
  const float* edge_feats = (const float*)d_in[1];
  const int* eidx = (const int*)d_in[2];  // [2, E]
  const int* batch = (const int*)d_in[3];
  // params: jax sorted-dict-key flat order
  const float* epW = (const float*)d_in[4];
  const float* epb = (const float*)d_in[5];
  const float* epbe = (const float*)d_in[6];
  const float* epg = (const float*)d_in[7];
  const float *lW1[3], *lW2[3], *lb1[3], *lb2[3], *lbe1[3], *lbe2[3], *lg1[3], *lg2[3];
  for (int l = 0; l < 3; l++) {
    int b = 8 + l * 8;
    lW1[l] = (const float*)d_in[b + 0];
    lW2[l] = (const float*)d_in[b + 1];
    lb1[l] = (const float*)d_in[b + 2];
    lb2[l] = (const float*)d_in[b + 3];
    lbe1[l] = (const float*)d_in[b + 4];
    lbe2[l] = (const float*)d_in[b + 5];
    lg1[l] = (const float*)d_in[b + 6];
    lg2[l] = (const float*)d_in[b + 7];
  }
  const float* npW = (const float*)d_in[32];
  const float* npb = (const float*)d_in[33];
  const float* npbe = (const float*)d_in[34];
  const float* npg = (const float*)d_in[35];
  const float* nbe = (const float*)d_in[36];
  const float* ng = (const float*)d_in[37];

  const int* e_row = eidx;            // source nodes
  const int* e_col = eidx + N_EDGES;  // destination nodes

  // workspace carve
  char* ws = (char*)d_ws;
  size_t off = 0;
  auto carve = [&](size_t bytes) -> void* {
    void* p = ws + off;
    off = (off + bytes + 255) & ~(size_t)255;
    return p;
  };
  float* x = (float*)carve((size_t)N_NODES * HID * 4);
  bf16_t* x_bf = (bf16_t*)carve((size_t)N_NODES * HID * 2);
  bf16_t* e_attr = (bf16_t*)carve((size_t)N_EDGES * HID * 2);
  float* msum = (float*)carve((size_t)N_NODES * HID * 4);
  float* cnt = (float*)carve((size_t)N_NODES * 4);
  float* gsum = (float*)carve((size_t)N_GRAPHS * HID * 4);
  float* gcnt = (float*)carve((size_t)N_GRAPHS * 4);
  bf16_t* Wnt = (bf16_t*)carve((size_t)HID * NODE_DIM * 2);
  bf16_t* Wet = (bf16_t*)carve((size_t)HID * EDGE_DIM * 2);
  bf16_t *W1t[3], *W2t[3];
  for (int l = 0; l < 3; l++) {
    W1t[l] = (bf16_t*)carve((size_t)HID * 2 * HID * 2);
    W2t[l] = (bf16_t*)carve((size_t)HID * HID * 2);
  }

  // weight prep (transpose + bf16)
  k_wt<<<(HID * NODE_DIM + 255) / 256, 256, 0, stream>>>(npW, Wnt, NODE_DIM, HID);
  k_wt<<<(HID * EDGE_DIM + 255) / 256, 256, 0, stream>>>(epW, Wet, EDGE_DIM, HID);
  for (int l = 0; l < 3; l++) {
    k_wt<<<(HID * 2 * HID + 255) / 256, 256, 0, stream>>>(lW1[l], W1t[l], 2 * HID, HID);
    k_wt<<<(HID * HID + 255) / 256, 256, 0, stream>>>(lW2[l], W2t[l], HID, HID);
  }

  // destination degrees (invariant across layers)
  hipMemsetAsync(cnt, 0, (size_t)N_NODES * 4, stream);
  k_degree<<<(N_EDGES + 255) / 256, 256, 0, stream>>>(e_col, cnt);

  // encoders
  const int node_tiles = N_NODES / 16, edge_tiles = N_EDGES / 16;
  k_proj<NODE_DIM><<<(node_tiles + 7) / 8, 256, 0, stream>>>(
      node_feats, N_NODES, Wnt, npb, npg, npbe, x, x_bf);
  k_proj<EDGE_DIM><<<(edge_tiles + 7) / 8, 256, 0, stream>>>(
      edge_feats, N_EDGES, Wet, epb, epg, epbe, nullptr, e_attr);

  // message-passing layers
  for (int l = 0; l < 3; l++) {
    hipMemsetAsync(msum, 0, (size_t)N_NODES * HID * 4, stream);
    k_msg<<<(edge_tiles + 7) / 8, 256, 0, stream>>>(
        x_bf, e_attr, e_row, e_col, W1t[l], lb1[l], lg1[l], lbe1[l], W2t[l],
        lb2[l], lg2[l], lbe2[l], msum);
    k_update<<<(N_NODES * HID + 255) / 256, 256, 0, stream>>>(x, x_bf, msum, cnt);
  }

  // final LN + graph pooling
  hipMemsetAsync(gsum, 0, (size_t)N_GRAPHS * HID * 4, stream);
  hipMemsetAsync(gcnt, 0, (size_t)N_GRAPHS * 4, stream);
  k_final<<<(N_NODES + 7) / 8, 256, 0, stream>>>(x, batch, ng, nbe, gsum, gcnt);
  k_out<<<(N_GRAPHS * HID + 255) / 256, 256, 0, stream>>>(gsum, gcnt, (float*)d_out);
}
